// CausalVideoAttention_17901423690131
// MI455X (gfx1250) — matrix operand order repaired and checked by reading, not compiled, not run on hardware
//
#include <hip/hip_runtime.h>
#include <math.h>

// Problem constants (from reference setup_inputs)
#define B_  2
#define S_  2048
#define D_  512
#define H_  8
#define DH_ 64
#define BS_ (B_ * S_)   // 4096 tokens

typedef __attribute__((ext_vector_type(16))) _Float16 v16h;
typedef __attribute__((ext_vector_type(8)))  _Float16 v8h;
typedef __attribute__((ext_vector_type(8)))  float    v8f;
typedef __attribute__((ext_vector_type(4)))  unsigned int u32x4;
typedef __attribute__((ext_vector_type(8)))  int      i32x8;
typedef __attribute__((ext_vector_type(4)))  int      i32x4;

static __device__ __forceinline__ v16h pack16(v8h lo, v8h hi) {
  return __builtin_shufflevector(lo, hi, 0,1,2,3,4,5,6,7,8,9,10,11,12,13,14,15);
}
static __device__ __forceinline__ v8f zero8() {
  v8f z = {0.f,0.f,0.f,0.f,0.f,0.f,0.f,0.f};
  return z;
}

// ---- WMMA fragment loaders (ISA 7.12.2 layouts, wave32) -------------------
static __device__ __forceinline__ v16h load_a_frag(const _Float16* A, int lda) {
  const int lane = threadIdx.x & 31;
  const int hlf  = lane >> 4;
  const int r    = lane & 15;
  const _Float16* p = A + (size_t)r * lda + hlf * 8;
  v8h lo = *(const v8h*)(p);        // K = half*8 .. +7
  v8h hi = *(const v8h*)(p + 16);   // K = 16+half*8 .. +7
  return pack16(lo, hi);
}
static __device__ __forceinline__ v16h load_bt_frag(const _Float16* Bt, int ldb) {
  const int lane  = threadIdx.x & 31;
  const int khalf = lane >> 4;
  const int n     = lane & 15;
  const _Float16* p = Bt + (size_t)n * ldb + khalf * 8;
  v8h lo = *(const v8h*)(p);
  v8h hi = *(const v8h*)(p + 16);
  return pack16(lo, hi);
}
static __device__ __forceinline__ void store_c(float* C, int ldc, v8f c) {
  const int lane = threadIdx.x & 31;
  const int hlf  = lane >> 4;
  const int n    = lane & 15;
  #pragma unroll
  for (int i = 0; i < 8; ++i) C[(size_t)(hlf * 8 + i) * ldc + n] = c[i];
}

// ---- fp32 -> fp16 convert --------------------------------------------------
__global__ void cvt_f32_to_f16_kernel(const float* __restrict__ in,
                                      _Float16* __restrict__ out, int n) {
  int i = blockIdx.x * blockDim.x + threadIdx.x;
  if (i < n) out[i] = (_Float16)in[i];
}

// ---- TDM: DMA a 64-row x 32-half 2D tile (row stride K halfs) into LDS ----
// LDS rows are hardware-padded: 16 DWORDs data + 4 DWORDs pad = 40-half rows
// (pad_interval=3 -> every 16 DW, pad_amount=3 -> 4 DW), matching BPAD below.
#define BPAD 40
#define HAVE_TDM (__has_builtin(__builtin_amdgcn_tensor_load_to_lds) && \
                  __has_builtin(__builtin_amdgcn_s_wait_tensorcnt))

#if HAVE_TDM
static __device__ __forceinline__ void tdm_load_b_tile(const _Float16* gsrc,
                                                       void* lds_dst, int k_stride) {
  const unsigned long long ga = (unsigned long long)(uintptr_t)gsrc;
  const unsigned int lds_off  = (unsigned int)(uintptr_t)lds_dst; // low 32 = LDS byte addr
  // D# group 0: count=1 | lds_addr | global_addr[56:0] | type=2
  u32x4 g0 = { 1u,
               lds_off,
               (unsigned int)(ga & 0xFFFFFFFFu),
               (unsigned int)((ga >> 32) & 0x01FFFFFFu) | (2u << 30) };
  // D# group 1: data_size=2B, pad_enable, pad_interval=16DW, pad_amount=4DW,
  // tensor_dim0=32, tensor_dim1=64 (== tile dims, no OOB), tile 32x64,
  // tensor_dim0_stride = k_stride elements.
  i32x8 g1 = { (int)((1u << 16) | (1u << 20) | (3u << 22) | (3u << 25)),
               (int)(32u << 16),          // tensor_dim0[15:0] in bits [31:16]
               (int)(64u << 16),          // dim0 hi=0 | tensor_dim1[15:0]
               (int)(32u << 16),          // dim1 hi=0 | tile_dim0=32
               (int)(64u),                // tile_dim1=64 | tile_dim2=0
               (int)k_stride,             // tensor_dim0_stride[31:0]
               0, 0 };
  i32x4 gz = {0, 0, 0, 0};
#if defined(__clang_major__) && (__clang_major__ >= 23)
  i32x8 gz8 = {0, 0, 0, 0, 0, 0, 0, 0};
  __builtin_amdgcn_tensor_load_to_lds(g0, g1, gz, gz, gz8, 0);
#else
  __builtin_amdgcn_tensor_load_to_lds(g0, g1, gz, gz, 0);
#endif
}
#endif

// ---- GEMM: C[M,N] = A[M,K] * B^T, A/Bt fp16 row-major, C fp32 -------------
// grid: (N/64, M/128); block 256 = 8 waves; wave w -> 16 rows, 64 cols.
// B tile (64x32) staged in LDS via TDM DMA (wave 0 issues, TENSORcnt fences),
// double buffered; compute overlaps the DMA of the next tile.
template <int K>
__global__ __launch_bounds__(256) void gemm_f16t_kernel(
    const _Float16* __restrict__ A, const _Float16* __restrict__ Bt,
    float* __restrict__ C, int N) {
  const int wave = threadIdx.x >> 5;                 // 0..7
  const int m0   = blockIdx.y * 128 + wave * 16;
  const int n0   = blockIdx.x * 64;

  __shared__ __align__(16) _Float16 Bs[2][64 * BPAD];

  const _Float16* Abase = A + (size_t)m0 * K;
  const _Float16* Btile = Bt + (size_t)n0 * K;       // rows n0.., stride K

#if HAVE_TDM
  if (threadIdx.x < 32) {                            // wave 0 DMAs tile 0
    tdm_load_b_tile(Btile, &Bs[0][0], K);
    __builtin_amdgcn_s_wait_tensorcnt(0);
  }
#else
  {
    const int lrow = threadIdx.x >> 2, lchk = threadIdx.x & 3;
    v8h g = *(const v8h*)(Btile + (size_t)lrow * K + lchk * 8);
    *(v8h*)(&Bs[0][lrow * BPAD + lchk * 8]) = g;
  }
#endif
  v16h a_buf[2];
  a_buf[0] = load_a_frag(Abase, K);
  __syncthreads();

  v8f acc[4] = {zero8(), zero8(), zero8(), zero8()};
  constexpr int NK = K / 32;
  #pragma unroll
  for (int kb = 0; kb < NK; ++kb) {
    // issue next tile's transfers first so they overlap this step's WMMAs
#if !HAVE_TDM
    v8h g;
#endif
    if (kb + 1 < NK) {
#if HAVE_TDM
      if (threadIdx.x < 32)
        tdm_load_b_tile(Btile + (kb + 1) * 32, &Bs[(kb + 1) & 1][0], K);
#else
      const int lrow = threadIdx.x >> 2, lchk = threadIdx.x & 3;
      g = *(const v8h*)(Btile + (size_t)lrow * K + (kb + 1) * 32 + lchk * 8);
#endif
      a_buf[(kb + 1) & 1] = load_a_frag(Abase + (kb + 1) * 32, K);
    }
    // batch all fragment ds_loads, then a back-to-back WMMA burst
    const _Float16* bs = &Bs[kb & 1][0];
    v16h bf[4];
    #pragma unroll
    for (int t = 0; t < 4; ++t)
      bf[t] = load_bt_frag(bs + (size_t)t * 16 * BPAD, BPAD);
    #pragma unroll
    for (int t = 0; t < 4; ++t)
      acc[t] = __builtin_amdgcn_wmma_f32_16x16x32_f16(false, a_buf[kb & 1], false,
                                                      bf[t], (short)0, acc[t],
                                                      false, false);
    if (kb + 1 < NK) {
#if HAVE_TDM
      if (threadIdx.x < 32) __builtin_amdgcn_s_wait_tensorcnt(0);
#else
      const int lrow = threadIdx.x >> 2, lchk = threadIdx.x & 3;
      *(v8h*)(&Bs[(kb + 1) & 1][lrow * BPAD + lchk * 8]) = g;
#endif
      __syncthreads();
    }
  }
  #pragma unroll
  for (int t = 0; t < 4; ++t)
    store_c(C + (size_t)m0 * N + n0 + t * 16, N, acc[t]);
}

// ---- RMSNorm(q,k) + split into head-major fp16 [b*h][s][dh] ---------------
__global__ __launch_bounds__(256) void rmsnorm_split_kernel(
    const float* __restrict__ qkv, const float* __restrict__ scale_q,
    const float* __restrict__ scale_k, _Float16* __restrict__ qh,
    _Float16* __restrict__ kh, _Float16* __restrict__ vh) {
  const int gwave = (blockIdx.x * blockDim.x + threadIdx.x) >> 5;
  const int lane  = threadIdx.x & 31;
  const int h   = gwave % H_;
  const int tok = gwave / H_;       // bi*S + si
  const int bi  = tok / S_;
  const int si  = tok % S_;
  const float* base = qkv + (size_t)tok * (3 * D_);
  const float* qp = base + h * DH_;
  const float* kp = base + D_ + h * DH_;
  const float* vp = base + 2 * D_ + h * DH_;
  float q0 = qp[lane], q1 = qp[lane + 32];
  float k0 = kp[lane], k1 = kp[lane + 32];
  float v0 = vp[lane], v1 = vp[lane + 32];
  float sq = q0 * q0 + q1 * q1;
  float sk = k0 * k0 + k1 * k1;
  #pragma unroll
  for (int off = 16; off >= 1; off >>= 1) {
    sq += __shfl_xor(sq, off, 32);
    sk += __shfl_xor(sk, off, 32);
  }
  const float rq = 1.0f / (sqrtf(sq * (1.0f / 64.0f)) + 1e-6f);
  const float rk = 1.0f / (sqrtf(sk * (1.0f / 64.0f)) + 1e-6f);
  const size_t o = ((size_t)(bi * H_ + h) * S_ + si) * DH_;
  qh[o + lane]      = (_Float16)(q0 * rq * scale_q[lane]);
  qh[o + lane + 32] = (_Float16)(q1 * rq * scale_q[lane + 32]);
  kh[o + lane]      = (_Float16)(k0 * rk * scale_k[lane]);
  kh[o + lane + 32] = (_Float16)(k1 * rk * scale_k[lane + 32]);
  vh[o + lane]      = (_Float16)v0;
  vh[o + lane + 32] = (_Float16)v1;
}

// ---- 64x64 tile transpose: V [bh][s][dh] -> Vt [bh][dh][S] ----------------
__global__ __launch_bounds__(256) void transpose_v_kernel(
    const _Float16* __restrict__ Vin, _Float16* __restrict__ Vt) {
  const int bh  = blockIdx.y;
  const int s0  = blockIdx.x * 64;
  const int row = threadIdx.x >> 2;   // 0..63
  const int chk = threadIdx.x & 3;    // 0..3
  __shared__ __align__(16) _Float16 tile[64][72];
  const _Float16* src = Vin + ((size_t)bh * S_ + s0 + row) * DH_;
  *(v8h*)(&tile[row][chk * 8])       = *(const v8h*)(src + chk * 8);
  *(v8h*)(&tile[row][(chk + 4) * 8]) = *(const v8h*)(src + (chk + 4) * 8);
  __syncthreads();
  #pragma unroll
  for (int c = 0; c < 2; ++c) {
    const int kc = chk + c * 4;       // key chunk 0..7
    v8h w;
    #pragma unroll
    for (int i = 0; i < 8; ++i) w[i] = tile[kc * 8 + i][row];
    *(v8h*)(Vt + ((size_t)bh * DH_ + row) * S_ + s0 + kc * 8) = w;
  }
}

// ---- Flash attention: causal, online softmax, all-WMMA, pipelined ---------
__global__ __launch_bounds__(128) void flash_attn_kernel(
    const _Float16* __restrict__ Q, const _Float16* __restrict__ K,
    const _Float16* __restrict__ Vt, _Float16* __restrict__ Z) {
  const int wq   = threadIdx.x >> 5;   // 0..3
  const int lane = threadIdx.x & 31;
  const int hlf  = lane >> 4;
  const int n16  = lane & 15;
  const int q0   = blockIdx.x * 64;
  const int bh   = blockIdx.y;         // bi*H + h
  const int bi   = bh / H_;
  const int h    = bh % H_;

  const _Float16* Qb  = Q  + (size_t)bh * S_ * DH_;
  const _Float16* Kb  = K  + (size_t)bh * S_ * DH_;
  const _Float16* Vtb = Vt + (size_t)bh * DH_ * S_;   // [dh][S]

  __shared__ __align__(16) _Float16 Pl[4][16][72];

  v16h qa0 = load_a_frag(Qb + (size_t)(q0 + wq * 16) * DH_ + 0,  DH_);
  v16h qa1 = load_a_frag(Qb + (size_t)(q0 + wq * 16) * DH_ + 32, DH_);

  float mrow[8], lrow[8];
  #pragma unroll
  for (int i = 0; i < 8; ++i) { mrow[i] = -1e30f; lrow[i] = 0.f; }
  v8f o[4] = {zero8(), zero8(), zero8(), zero8()};

  v16h kbf[8];
  #pragma unroll
  for (int nt = 0; nt < 4; ++nt) {
    kbf[nt * 2 + 0] = load_bt_frag(Kb + (size_t)(nt * 16) * DH_ + 0,  DH_);
    kbf[nt * 2 + 1] = load_bt_frag(Kb + (size_t)(nt * 16) * DH_ + 32, DH_);
  }

  for (int kt = 0; kt <= q0; kt += 64) {
    // ---- S = Q * K^T  (16 x 64), 8 WMMAs on resident fragments
    v8f s[4] = {zero8(), zero8(), zero8(), zero8()};
    #pragma unroll
    for (int nt = 0; nt < 4; ++nt) {
      s[nt] = __builtin_amdgcn_wmma_f32_16x16x32_f16(false, qa0, false, kbf[nt*2+0],
                                                     (short)0, s[nt], false, false);
      s[nt] = __builtin_amdgcn_wmma_f32_16x16x32_f16(false, qa1, false, kbf[nt*2+1],
                                                     (short)0, s[nt], false, false);
    }
    // ---- issue V loads for this tile first (in-order loadcnt retirement
    //      lets the P@V WMMAs wait only on these, not on next-K loads)
    v16h vbf[8];
    #pragma unroll
    for (int nt = 0; nt < 4; ++nt) {
      vbf[nt * 2 + 0] = load_bt_frag(Vtb + (size_t)(nt * 16) * S_ + kt + 0,  S_);
      vbf[nt * 2 + 1] = load_bt_frag(Vtb + (size_t)(nt * 16) * S_ + kt + 32, S_);
    }
    // ---- then issue K loads for the next tile
    if (kt < q0) {
      #pragma unroll
      for (int nt = 0; nt < 4; ++nt) {
        kbf[nt*2+0] = load_bt_frag(Kb + (size_t)(kt + 64 + nt * 16) * DH_ + 0,  DH_);
        kbf[nt*2+1] = load_bt_frag(Kb + (size_t)(kt + 64 + nt * 16) * DH_ + 32, DH_);
      }
      if (kt + 128 <= q0)   // L2 prefetch for the tile after next
        __builtin_prefetch(Kb + (size_t)(kt + 128) * DH_, 0, 3);
    }
    // ---- causal mask: only the diagonal tile (BQ == BK == 64)
    if (kt == q0) {
      #pragma unroll
      for (int nt = 0; nt < 4; ++nt)
        #pragma unroll
        for (int i = 0; i < 8; ++i) {
          const int key = kt + nt * 16 + n16;
          const int row = q0 + wq * 16 + hlf * 8 + i;
          if (key > row) s[nt][i] = -1e30f;
        }
    }
    // ---- online softmax (VALU-heavy; overlaps the loads issued above)
    float mt[8];
    #pragma unroll
    for (int i = 0; i < 8; ++i) {
      float v = fmaxf(fmaxf(s[0][i], s[1][i]), fmaxf(s[2][i], s[3][i]));
      #pragma unroll
      for (int off = 8; off >= 1; off >>= 1)
        v = fmaxf(v, __shfl_xor(v, off, 32));
      mt[i] = v;
    }
    float alpha[8];
    #pragma unroll
    for (int i = 0; i < 8; ++i) {
      const float mn = fmaxf(mrow[i], mt[i]);
      alpha[i] = __expf(mrow[i] - mn);
      mrow[i]  = mn;
    }
    float psum[8];
    #pragma unroll
    for (int i = 0; i < 8; ++i) psum[i] = 0.f;
    #pragma unroll
    for (int nt = 0; nt < 4; ++nt)
      #pragma unroll
      for (int i = 0; i < 8; ++i) {
        const float p = __expf(s[nt][i] - mrow[i]);
        psum[i] += p;
        Pl[wq][hlf * 8 + i][nt * 16 + n16] = (_Float16)p;
      }
    #pragma unroll
    for (int i = 0; i < 8; ++i) {
      float v = psum[i];
      #pragma unroll
      for (int off = 8; off >= 1; off >>= 1)
        v += __shfl_xor(v, off, 32);
      lrow[i] = alpha[i] * lrow[i] + v;
    }
    #pragma unroll
    for (int nt = 0; nt < 4; ++nt)
      #pragma unroll
      for (int i = 0; i < 8; ++i) o[nt][i] = o[nt][i] * alpha[i];

    // ---- P back as A fragments (wave-private LDS round trip)
    v16h pa0 = load_a_frag(&Pl[wq][0][0], 72);
    v16h pa1 = load_a_frag(&Pl[wq][0][0] + 32, 72);
    // ---- O += P * V (V fragments already in flight/resident)
    #pragma unroll
    for (int nt = 0; nt < 4; ++nt) {
      o[nt] = __builtin_amdgcn_wmma_f32_16x16x32_f16(false, pa0, false, vbf[nt*2+0],
                                                     (short)0, o[nt], false, false);
      o[nt] = __builtin_amdgcn_wmma_f32_16x16x32_f16(false, pa1, false, vbf[nt*2+1],
                                                     (short)0, o[nt], false, false);
    }
  }

  // ---- epilogue: O /= l, write z[b, row, h*dh + col] as fp16 for out-proj
  #pragma unroll
  for (int i = 0; i < 8; ++i) lrow[i] = 1.0f / lrow[i];
  _Float16* Zb = Z + (size_t)bi * S_ * D_ + (size_t)h * DH_;
  #pragma unroll
  for (int nt = 0; nt < 4; ++nt)
    #pragma unroll
    for (int i = 0; i < 8; ++i) {
      const int row = q0 + wq * 16 + hlf * 8 + i;
      Zb[(size_t)row * D_ + nt * 16 + n16] = (_Float16)(o[nt][i] * lrow[i]);
    }
}

// ---------------------------------------------------------------------------
extern "C" void kernel_launch(void* const* d_in, const int* in_sizes, int n_in,
                              void* d_out, int out_size, void* d_ws, size_t ws_size,
                              hipStream_t stream) {
  (void)in_sizes; (void)n_in; (void)out_size; (void)ws_size;
  const float* x       = (const float*)d_in[0];   // [2,2048,512]
  const float* Wqkv    = (const float*)d_in[1];   // [1536,512]
  const float* Wo      = (const float*)d_in[2];   // [512,512]
  const float* scale_q = (const float*)d_in[3];   // [64]
  const float* scale_k = (const float*)d_in[4];   // [64]
  // d_in[5] = causal mask; computed analytically in-kernel.
  float* out = (float*)d_out;
  char*  ws  = (char*)d_ws;

  // workspace map (all regions fully written before read; ~47 MiB used)
  _Float16* xh    = (_Float16*)(ws + (size_t) 0 * (1u << 20));  // 4   MiB
  _Float16* wqkvh = (_Float16*)(ws + (size_t) 4 * (1u << 20));  // 1.5 MiB
  _Float16* woh   = (_Float16*)(ws + (size_t) 6 * (1u << 20));  // 0.5 MiB
  float*    qkvf  = (float*)   (ws + (size_t) 7 * (1u << 20));  // 24  MiB
  _Float16* qh    = (_Float16*)(ws + (size_t)31 * (1u << 20));  // 4   MiB
  _Float16* kh    = (_Float16*)(ws + (size_t)35 * (1u << 20));  // 4   MiB
  _Float16* vh    = (_Float16*)(ws + (size_t)39 * (1u << 20));  // 4   MiB
  _Float16* vt    = (_Float16*)(ws + (size_t)43 * (1u << 20));  // 4   MiB
  _Float16* zh    = xh;  // x fp16 no longer needed after QKV GEMM -> reuse

  // 1) fp32 -> fp16
  {
    int n = BS_ * D_;
    cvt_f32_to_f16_kernel<<<dim3((n + 255) / 256), 256, 0, stream>>>(x, xh, n);
    n = 3 * D_ * D_;
    cvt_f32_to_f16_kernel<<<dim3((n + 255) / 256), 256, 0, stream>>>(Wqkv, wqkvh, n);
    n = D_ * D_;
    cvt_f32_to_f16_kernel<<<dim3((n + 255) / 256), 256, 0, stream>>>(Wo, woh, n);
  }
  // 2) qkv = x @ Wqkv^T   [4096 x 1536 x 512]
  gemm_f16t_kernel<D_><<<dim3((3 * D_) / 64, BS_ / 128), 256, 0, stream>>>(
      xh, wqkvh, qkvf, 3 * D_);
  // 3) RMSNorm(q,k) + head split to fp16
  rmsnorm_split_kernel<<<dim3((BS_ * H_) / 8), 256, 0, stream>>>(
      qkvf, scale_q, scale_k, qh, kh, vh);
  // 3b) transpose V for contiguous P@V fragments
  transpose_v_kernel<<<dim3(S_ / 64, B_ * H_), 256, 0, stream>>>(vh, vt);
  // 4) causal flash attention (writes zh; zh aliases xh — safe, step 2
  //    consumed xh entirely before this kernel runs)
  flash_attn_kernel<<<dim3(S_ / 64, B_ * H_), 128, 0, stream>>>(qh, kh, vt, zh);
  // 5) out = z @ Wo^T   [4096 x 512 x 512]
  gemm_f16t_kernel<D_><<<dim3(D_ / 64, BS_ / 128), 256, 0, stream>>>(
      zh, woh, out, D_);
}
